// Encoder_6622839571237
// MI455X (gfx1250) — matrix operand order repaired and checked by reading, not compiled, not run on hardware
//
#include <hip/hip_runtime.h>
#include <hip/hip_bf16.h>

typedef _Float16 half_t;
typedef __attribute__((ext_vector_type(16))) _Float16 v16h;
typedef __attribute__((ext_vector_type(8)))  float    v8f;
typedef unsigned int u32;

#define ENC_B   2
#define ENC_S   2048
#define ENC_D   512
#define ENC_H   8
#define ENC_DK  64
#define ENC_L   6
#define ENC_DFF 2048
#define ENC_M   (ENC_B * ENC_S)   // 4096 tokens

// ---------------------------------------------------------------------------
// fp32 -> f16 transposed weight conversion: Wt[n*K + k] = W[k*N + n]
// ---------------------------------------------------------------------------
__global__ __launch_bounds__(256)
void convert_weight_t(const float* __restrict__ W, half_t* __restrict__ Wt,
                      int K, int N) {
  int i = blockIdx.x * 256 + threadIdx.x;
  if (i >= K * N) return;
  int n = i / K;
  int k = i - n * K;
  Wt[(size_t)n * K + k] = (half_t)W[(size_t)k * N + n];
}

// ---------------------------------------------------------------------------
// x[b,s,d] = emb[src[b,s], d] * sqrt(D) + pe[s, d]
// ---------------------------------------------------------------------------
__global__ __launch_bounds__(256)
void embed_pe(const int* __restrict__ src, const float* __restrict__ emb,
              const float* __restrict__ pe, float* __restrict__ x) {
  int i = blockIdx.x * 256 + threadIdx.x;       // over B*S*D
  int d   = i & (ENC_D - 1);
  int tok = i >> 9;                              // D = 512 = 2^9
  int s   = tok & (ENC_S - 1);
  x[i] = emb[(size_t)src[tok] * ENC_D + d] * 22.62741699796952f
       + pe[(size_t)s * ENC_D + d];
}

// ---------------------------------------------------------------------------
// Pre-LN layernorm: one wave32 per row of 512. Writes f16 or f32.
// ---------------------------------------------------------------------------
template <bool OUT16>
__global__ __launch_bounds__(256)
void layernorm_rows(const float* __restrict__ x,
                    const float* __restrict__ g, const float* __restrict__ b,
                    half_t* __restrict__ o16, float* __restrict__ o32) {
  const int lane = threadIdx.x & 31;
  const int row  = blockIdx.x * 8 + (threadIdx.x >> 5);
  const float* xr = x + (size_t)row * ENC_D;

  float v[16];
  float s = 0.f;
#pragma unroll
  for (int i = 0; i < 16; ++i) { v[i] = xr[lane + i * 32]; s += v[i]; }
#pragma unroll
  for (int o = 16; o; o >>= 1) s += __shfl_xor(s, o, 32);
  const float mu = s * (1.f / (float)ENC_D);

  float q = 0.f;
#pragma unroll
  for (int i = 0; i < 16; ++i) { float d = v[i] - mu; q += d * d; }
#pragma unroll
  for (int o = 16; o; o >>= 1) q += __shfl_xor(q, o, 32);
  const float rstd = rsqrtf(q * (1.f / (float)ENC_D) + 1e-5f);

#pragma unroll
  for (int i = 0; i < 16; ++i) {
    int c = lane + i * 32;
    float y = (v[i] - mu) * rstd * g[c] + b[c];
    if (OUT16) o16[(size_t)row * ENC_D + c] = (half_t)y;
    else       o32[(size_t)row * ENC_D + c] = y;
  }
}

// ---------------------------------------------------------------------------
// Masked softmax over one score row of S=2048 (block of 256, 8 elems/thread).
// Matches reference: -inf where key is PAD, nan_to_num -> 0 if row dead.
// ---------------------------------------------------------------------------
__global__ __launch_bounds__(256)
void softmax_mask(const float* __restrict__ scores, half_t* __restrict__ probs,
                  const int* __restrict__ srcb, float scale) {
  __shared__ float red[8];
  const int tid = threadIdx.x;
  const float* row = scores + (size_t)blockIdx.x * ENC_S;
  half_t* prow = probs + (size_t)blockIdx.x * ENC_S;

  float sv[8];
  float mx = -__builtin_inff();
#pragma unroll
  for (int i = 0; i < 8; ++i) {
    int j = tid + i * 256;
    float v = (srcb[j] != 0) ? row[j] * scale : -__builtin_inff();
    sv[i] = v;
    mx = fmaxf(mx, v);
  }
#pragma unroll
  for (int o = 16; o; o >>= 1) mx = fmaxf(mx, __shfl_xor(mx, o, 32));
  if ((tid & 31) == 0) red[tid >> 5] = mx;
  __syncthreads();
  float bmx = red[0];
#pragma unroll
  for (int k = 1; k < 8; ++k) bmx = fmaxf(bmx, red[k]);
  __syncthreads();

  const bool dead = !(bmx > -3.0e38f);   // whole row masked -> nan_to_num = 0
  float sum = 0.f;
#pragma unroll
  for (int i = 0; i < 8; ++i) {
    float e = dead ? 0.f : __expf(sv[i] - bmx);
    sv[i] = e;
    sum += e;
  }
#pragma unroll
  for (int o = 16; o; o >>= 1) sum += __shfl_xor(sum, o, 32);
  if ((tid & 31) == 0) red[tid >> 5] = sum;
  __syncthreads();
  float bsum = 0.f;
#pragma unroll
  for (int k = 0; k < 8; ++k) bsum += red[k];
  const float inv = (bsum > 0.f) ? (1.f / bsum) : 0.f;
#pragma unroll
  for (int i = 0; i < 8; ++i)
    prow[tid + i * 256] = (half_t)(sv[i] * inv);
}

// ---------------------------------------------------------------------------
// v[B,S,D] (f16) -> vT[B,H,DK,S] (f16) so PV GEMM's B-matrix is [N=DK][K=S].
// ---------------------------------------------------------------------------
__global__ __launch_bounds__(256)
void transpose_heads(const half_t* __restrict__ v, half_t* __restrict__ vT) {
  int i = blockIdx.x * 256 + threadIdx.x;        // over B*S*D
  int d = i & (ENC_D - 1);
  int s = (i >> 9) & (ENC_S - 1);
  int b = i >> 20;                               // S*D = 2^20
  int h = d >> 6, dk = d & 63;
  vT[((((size_t)b * ENC_H + h) * ENC_DK + dk) << 11) + s] = v[i];
}

// ---------------------------------------------------------------------------
// WMMA GEMM: C[M,N] = A[M,K](f16,lda) * Bt[N,K](f16,ldb) (+bias)(+residual)
//            optional ReLU; f32 and/or f16 stores with row stride ldc.
// Epilogue configuration is compile-time (template) -> branch-free epilogues.
//
// Block = 256 threads = 8 waves arranged 4(M) x 2(N); each wave owns a
// 32x32 output block = 2x2 WMMA tiles -> 4 v_wmma_f32_16x16x32_f16 per
// K-step with A/B fragment reuse. Macro-tile 128(M) x 64(N), K-step 32.
// LDS is double-buffered: one s_barrier per K-step; next chunk's
// global_load_b128s issue before the current chunk's WMMAs so VMEM latency
// overlaps matrix work; global_prefetch_b8 targets chunk+2.
//
// Fragment layouts per CDNA5 ISA 7.12.2:
//   A 16x32 f16 : lane%16 = M, lane/16 = K-half; VGPR i -> K pair
//   B 32x16 f16 : lane%16 = N, lane/16 = K-half (k 8h+0..7, 16+8h+0..7); VGPR halves = K
//   C/D 16x16   : lane%16 = N, VGPR r -> M = r + (lane/16)*8
// ---------------------------------------------------------------------------
#define BM 128
#define BN 64
#define TK 32
#define LDS_STRIDE 40   // 40 halves = 80B rows: 16B-aligned, bank-spread

union FragA { v16h h; u32  u[8]; };
union FragB { v16h h; uint4 q[2]; };

template <bool HAS_BIAS, bool HAS_RES, bool OUT32, bool OUT16, bool RELU>
__global__ __launch_bounds__(256)
void gemm_f16_wmma(const half_t* __restrict__ A, int lda,
                   const half_t* __restrict__ Bt, int ldb, int K,
                   const float* __restrict__ bias,
                   const float* __restrict__ residual,
                   float* __restrict__ C32, half_t* __restrict__ C16,
                   int ldc) {
  __shared__ __attribute__((aligned(16))) half_t lsA[2][BM][LDS_STRIDE];
  __shared__ __attribute__((aligned(16))) half_t lsB[2][BN][LDS_STRIDE];

  const int tid  = threadIdx.x;
  const int lane = tid & 31;
  const int wave = tid >> 5;
  const int wm = wave & 3;           // 4 wave-rows of 32 along M
  const int wn = wave >> 2;          // 2 wave-cols of 32 along N

  // cooperative staging: A 128x32 (2 x uint4/thread), B 64x32 (1 x uint4)
  const int arow = tid >> 1, aseg = (tid & 1) * 16;  // 128 rows x 2 segs of 16
  const int brow = tid >> 2, bseg = (tid & 3) * 8;   // 64 rows x 4 segs of 8
  const half_t* gA = A  + (size_t)(blockIdx.y * BM + arow) * lda + aseg;
  const half_t* gB = Bt + (size_t)(blockIdx.x * BN + brow) * ldb + bseg;

  const int nl = lane & 15;          // N within tile (also M for A-frag)
  const int hs = lane >> 4;          // K-half / K-group selector

  v8f acc[2][2] = {};

  // prologue: stage chunk 0 into buffer 0
  {
    uint4 a0 = *(const uint4*)(gA);
    uint4 a1 = *(const uint4*)(gA + 8);
    uint4 b0 = *(const uint4*)(gB);
    *(uint4*)&lsA[0][arow][aseg]     = a0;
    *(uint4*)&lsA[0][arow][aseg + 8] = a1;
    *(uint4*)&lsB[0][brow][bseg]     = b0;
  }
  __syncthreads();

  int buf = 0;
  for (int kk = 0; kk < K; kk += TK) {
    const bool more = (kk + TK) < K;
    uint4 a0, a1, b0;
    if (more) {                       // issue next chunk's loads early
      a0 = *(const uint4*)(gA + kk + TK);
      a1 = *(const uint4*)(gA + kk + TK + 8);
      b0 = *(const uint4*)(gB + kk + TK);
      if (kk + 2 * TK < K) {          // gfx1250 global_prefetch_b8, chunk+2
        __builtin_prefetch(gA + kk + 2 * TK, 0, 1);
        __builtin_prefetch(gB + kk + 2 * TK, 0, 1);
      }
    }

    // fragments from current buffer
    FragA fa[2];
    FragB fb[2];
#pragma unroll
    for (int mi = 0; mi < 2; ++mi) {
      const u32* ap = (const u32*)&lsA[buf][wm * 32 + mi * 16 + nl][0];
#pragma unroll
      for (int i = 0; i < 8; ++i) {
        int dw = ((i < 4) ? 0 : 8) + hs * 4 + (i & 3);  // K-pair index
        fa[mi].u[i] = ap[dw];
      }
    }
#pragma unroll
    for (int ni = 0; ni < 2; ++ni) {
      const uint4* bp = (const uint4*)&lsB[buf][wn * 32 + ni * 16 + nl][hs * 8];
      fb[ni].q[0] = bp[0];
      fb[ni].q[1] = bp[2];
    }

#pragma unroll
    for (int mi = 0; mi < 2; ++mi)
#pragma unroll
      for (int ni = 0; ni < 2; ++ni)
        acc[mi][ni] = __builtin_amdgcn_wmma_f32_16x16x32_f16(
            false, fa[mi].h, false, fb[ni].h, (short)0, acc[mi][ni],
            false, false);

    if (more) {                       // store next chunk into other buffer
      *(uint4*)&lsA[buf ^ 1][arow][aseg]     = a0;
      *(uint4*)&lsA[buf ^ 1][arow][aseg + 8] = a1;
      *(uint4*)&lsB[buf ^ 1][brow][bseg]     = b0;
    }
    __syncthreads();
    buf ^= 1;
  }

  // branch-free epilogue (config compile-time)
#pragma unroll
  for (int mi = 0; mi < 2; ++mi) {
#pragma unroll
    for (int ni = 0; ni < 2; ++ni) {
#pragma unroll
      for (int r = 0; r < 8; ++r) {
        int row = blockIdx.y * BM + wm * 32 + mi * 16 + r + hs * 8;
        int col = blockIdx.x * BN + wn * 32 + ni * 16 + nl;
        size_t off = (size_t)row * ldc + col;
        float v = acc[mi][ni][r];
        if (HAS_BIAS) v += bias[col];
        if (HAS_RES)  v += residual[off];
        if (RELU)     v = v > 0.f ? v : 0.f;
        if (OUT32) C32[off] = v;
        if (OUT16) C16[off] = (half_t)v;
      }
    }
  }
}

// instantiation aliases        bias   res    f32    f16    relu
#define GEMM_F16   (gemm_f16_wmma<false, false, false, true,  false>)  // QKV, PV
#define GEMM_F32   (gemm_f16_wmma<false, false, true,  false, false>)  // scores
#define GEMM_RES32 (gemm_f16_wmma<false, true,  true,  false, false>)  // WO proj
#define GEMM_FFN1  (gemm_f16_wmma<true,  false, false, true,  true >)  // FFN1
#define GEMM_FFN2  (gemm_f16_wmma<true,  true,  true,  false, false>)  // FFN2

// ---------------------------------------------------------------------------
// Host orchestration
// ---------------------------------------------------------------------------
extern "C" void kernel_launch(void* const* d_in, const int* in_sizes, int n_in,
                              void* d_out, int out_size, void* d_ws, size_t ws_size,
                              hipStream_t stream) {
  const int B = ENC_B, S = ENC_S, D = ENC_D, H = ENC_H, DK = ENC_DK;
  const int L = ENC_L, DFF = ENC_DFF, M = ENC_M;

  const int*   src   = (const int*)  d_in[0];
  const float* emb   = (const float*)d_in[1];
  const float* pe    = (const float*)d_in[2];
  const float* WQ    = (const float*)d_in[3];
  const float* WK    = (const float*)d_in[4];
  const float* WV    = (const float*)d_in[5];
  const float* WO    = (const float*)d_in[6];
  const float* W1    = (const float*)d_in[7];
  const float* b1    = (const float*)d_in[8];
  const float* W2    = (const float*)d_in[9];
  const float* b2    = (const float*)d_in[10];
  const float* ln1_s = (const float*)d_in[11];
  const float* ln1_b = (const float*)d_in[12];
  const float* ln2_s = (const float*)d_in[13];
  const float* ln2_b = (const float*)d_in[14];
  const float* fn_s  = (const float*)d_in[15];
  const float* fn_b  = (const float*)d_in[16];

  // ---- carve workspace (256B aligned) ----
  size_t woff = 0;
  auto carve = [&](size_t bytes) -> void* {
    void* p = (char*)d_ws + woff;
    woff += (bytes + 255) & ~(size_t)255;
    return p;
  };
  half_t* WQt = (half_t*)carve((size_t)L * D * D   * 2);
  half_t* WKt = (half_t*)carve((size_t)L * D * D   * 2);
  half_t* WVt = (half_t*)carve((size_t)L * D * D   * 2);
  half_t* WOt = (half_t*)carve((size_t)L * D * D   * 2);
  half_t* W1t = (half_t*)carve((size_t)L * D * DFF * 2);
  half_t* W2t = (half_t*)carve((size_t)L * DFF * D * 2);
  float*  x      = (float*) carve((size_t)M * D * 4);
  half_t* h16    = (half_t*)carve((size_t)M * D * 2);
  half_t* q16    = (half_t*)carve((size_t)M * D * 2);
  half_t* k16    = (half_t*)carve((size_t)M * D * 2);
  half_t* v16    = (half_t*)carve((size_t)M * D * 2);
  half_t* vT     = (half_t*)carve((size_t)M * D * 2);          // [B,H,DK,S]
  half_t* attn16 = (half_t*)carve((size_t)M * D * 2);
  half_t* h1_16  = (half_t*)carve((size_t)M * DFF * 2);
  float*  scores = (float*) carve((size_t)S * S * 4);          // one (b,h) at a time
  half_t* probs  = (half_t*)carve((size_t)S * S * 2);

  // ---- one-time weight conversion (transposed f16) ----
  for (int l = 0; l < L; ++l) {
    size_t dd = (size_t)l * D * D;
    size_t df = (size_t)l * D * DFF;
    convert_weight_t<<<(D * D + 255) / 256, 256, 0, stream>>>(WQ + dd, WQt + dd, D, D);
    convert_weight_t<<<(D * D + 255) / 256, 256, 0, stream>>>(WK + dd, WKt + dd, D, D);
    convert_weight_t<<<(D * D + 255) / 256, 256, 0, stream>>>(WV + dd, WVt + dd, D, D);
    convert_weight_t<<<(D * D + 255) / 256, 256, 0, stream>>>(WO + dd, WOt + dd, D, D);
    convert_weight_t<<<(D * DFF + 255) / 256, 256, 0, stream>>>(W1 + df, W1t + df, D, DFF);
    convert_weight_t<<<(D * DFF + 255) / 256, 256, 0, stream>>>(W2 + df, W2t + df, DFF, D);
  }

  // ---- embedding + positional encoding ----
  embed_pe<<<(M * D) / 256, 256, 0, stream>>>(src, emb, pe, x);

  const dim3 blk(256);
  const float inv_sqrt_dk = 0.125f;   // 1/sqrt(64)

  for (int l = 0; l < L; ++l) {
    size_t dd = (size_t)l * D * D;
    size_t df = (size_t)l * D * DFF;

    // ---- pre-LN attention ----
    layernorm_rows<true><<<M / 8, blk, 0, stream>>>(
        x, ln1_s + l * D, ln1_b + l * D, h16, nullptr);
    GEMM_F16<<<dim3(D / BN, M / BM), blk, 0, stream>>>(
        h16, D, WQt + dd, D, D, nullptr, nullptr, nullptr, q16, D);
    GEMM_F16<<<dim3(D / BN, M / BM), blk, 0, stream>>>(
        h16, D, WKt + dd, D, D, nullptr, nullptr, nullptr, k16, D);
    GEMM_F16<<<dim3(D / BN, M / BM), blk, 0, stream>>>(
        h16, D, WVt + dd, D, D, nullptr, nullptr, nullptr, v16, D);
    transpose_heads<<<(M * D) / 256, blk, 0, stream>>>(v16, vT);

    for (int b = 0; b < B; ++b) {
      for (int h = 0; h < H; ++h) {
        const half_t* qh = q16 + (size_t)b * S * D + h * DK;
        const half_t* kh = k16 + (size_t)b * S * D + h * DK;   // [N=S][K=DK]
        const half_t* vh = vT + ((size_t)b * H + h) * DK * S;  // [N=DK][K=S]
        half_t* oh = attn16 + (size_t)b * S * D + h * DK;

        // scores = q @ k^T   (M=S, N=S, K=DK)
        GEMM_F32<<<dim3(S / BN, S / BM), blk, 0, stream>>>(
            qh, D, kh, D, DK, nullptr, nullptr, scores, nullptr, S);
        // masked softmax (scale folded in), -> f16 probs
        softmax_mask<<<S, blk, 0, stream>>>(scores, probs, src + b * S,
                                            inv_sqrt_dk);
        // attn = probs @ v   (M=S, N=DK, K=S), strided f16 write into [B,S,D]
        GEMM_F16<<<dim3(DK / BN, S / BM), blk, 0, stream>>>(
            probs, S, vh, S, S, nullptr, nullptr, nullptr, oh, D);
      }
    }
    // x = x + attn @ WO
    GEMM_RES32<<<dim3(D / BN, M / BM), blk, 0, stream>>>(
        attn16, D, WOt + dd, D, D, nullptr, x, x, nullptr, D);

    // ---- pre-LN FFN ----
    layernorm_rows<true><<<M / 8, blk, 0, stream>>>(
        x, ln2_s + l * D, ln2_b + l * D, h16, nullptr);
    GEMM_FFN1<<<dim3(DFF / BN, M / BM), blk, 0, stream>>>(
        h16, D, W1t + df, D, D, b1 + (size_t)l * DFF, nullptr,
        nullptr, h1_16, DFF);
    GEMM_FFN2<<<dim3(D / BN, M / BM), blk, 0, stream>>>(
        h1_16, DFF, W2t + df, DFF, DFF, b2 + (size_t)l * D, x, x, nullptr, D);
  }

  // ---- final layernorm -> fp32 output ----
  layernorm_rows<false><<<M / 8, blk, 0, stream>>>(
      x, fn_s, fn_b, nullptr, (float*)d_out);
}
